// SwinTransformerBlock_50225347559868
// MI455X (gfx1250) — matrix operand order repaired and checked by reading, not compiled, not run on hardware
//
#include <hip/hip_runtime.h>
#include <math.h>

// ---------------- CDNA5 WMMA types ----------------
typedef __attribute__((ext_vector_type(16))) _Float16 v16h;
typedef __attribute__((ext_vector_type(8)))  _Float16 v8h;
typedef __attribute__((ext_vector_type(8)))  float    v8f;

#define DIMC   512
#define NHEAD  4
#define HD     128           // head dim
#define WS_    8
#define NTOK   64            // tokens per window
#define BATCH  16
#define SS_    4
#define NWIN   1024          // windows total
#define MROWS  65536         // tokens total
#define QSCALE 0.08838834764831845f              // 128^-0.5

// ======================================================================
// Fragment loads matching CDNA5 ISA WMMA VGPR layouts (wave32).
// A (16x32 f16, MxK): lane l: row = l&15; halves 0..7 = K[k0+(l>>4)*8 ..),
//                     halves 8..15 = K[k0+16+(l>>4)*8 ..)
// B (32x16 f16, KxN): lane l: col = l&15; halves 0..7 = K[k0+(l>>4)*8 ..),
//                     halves 8..15 = K[k0+16+(l>>4)*8 ..)
// C/D (16x16 f32):    lane l, vgpr r: M = r + 8*(l>>4), N = l&15
// ======================================================================
__device__ static inline v16h load_a_frag(const _Float16* A, int lda,
                                          int row0, int k0, int lane) {
    int ln = lane & 15, lg = lane >> 4;
    const _Float16* p = A + (size_t)(row0 + ln) * lda + k0 + lg * 8;
    v8h lo = *(const v8h*)p;          // K chunk 0  (16B)
    v8h hi = *(const v8h*)(p + 16);   // K chunk +16 (16B)
    return __builtin_shufflevector(lo, hi, 0,1,2,3,4,5,6,7,
                                           8,9,10,11,12,13,14,15);
}

// Bt is B transposed, row-major [N][K] -> a column of B is contiguous.
__device__ static inline v16h load_b_frag(const _Float16* Bt, int ldb,
                                          int col0, int k0, int lane) {
    int ln = lane & 15, lg = lane >> 4;
    const _Float16* p = Bt + (size_t)(col0 + ln) * ldb + k0 + lg * 8;
    v8h lo = *(const v8h*)p;
    v8h hi = *(const v8h*)(p + 16);
    return __builtin_shufflevector(lo, hi, 0,1,2,3,4,5,6,7,
                                           8,9,10,11,12,13,14,15);
}

__device__ static inline v8f wmma_f16(v16h a, v16h b, v8f c) {
    return __builtin_amdgcn_wmma_f32_16x16x32_f16(
        false, a, false, b, (short)0, c, false, false);
}

// Core: wave computes a 32(M) x (NT*16)(N) tile.
template<int NT>
__device__ static inline void gemm_core(const _Float16* A, int lda,
                                        const _Float16* Bt, int ldb,
                                        int K, int row0, int col0,
                                        int lane, v8f acc[2][NT]) {
    int ln = lane & 15, lg = lane >> 4;
    for (int k0 = 0; k0 < K; k0 += 32) {
        if (k0 + 32 < K) {   // WGP-scope prefetch of next A chunks
            __builtin_prefetch(A + (size_t)(row0 + ln) * lda + k0 + 32 + lg * 8, 0, 3);
            __builtin_prefetch(A + (size_t)(row0 + 16 + ln) * lda + k0 + 32 + lg * 8, 0, 3);
        }
        v16h a0 = load_a_frag(A, lda, row0,      k0, lane);
        v16h a1 = load_a_frag(A, lda, row0 + 16, k0, lane);
        #pragma unroll
        for (int ct = 0; ct < NT; ++ct) {
            v16h b = load_b_frag(Bt, ldb, col0 + ct * 16, k0, lane);
            acc[0][ct] = wmma_f16(a0, b, acc[0][ct]);
            acc[1][ct] = wmma_f16(a1, b, acc[1][ct]);
        }
    }
}

// ======================================================================
// Prep kernels
// ======================================================================
// Wt[n*K + k] = (f16) W[k*N + n]
__global__ void tpose_cvt_kernel(const float* __restrict__ W,
                                 _Float16* __restrict__ Wt, int K, int N) {
    int tid = blockIdx.x * blockDim.x + threadIdx.x;
    if (tid >= N * K) return;
    int n = tid / K, k = tid - n * K;
    Wt[tid] = (_Float16)W[(size_t)k * N + n];
}

// biasmat[n][m] = bias_table[rpi(n,m)]
__global__ void biasmat_kernel(const float* __restrict__ table,
                               float* __restrict__ biasmat) {
    int tid = blockIdx.x * blockDim.x + threadIdx.x;   // 4096
    int n = tid >> 6, m = tid & 63;
    int i0 = n >> 3, j0 = n & 7, i1 = m >> 3, j1 = m & 7;
    int idx = (i0 - i1 + WS_ - 1) * (2 * WS_ - 1) + (j0 - j1 + WS_ - 1);
    biasmat[tid] = table[idx];
}

// ======================================================================
// LayerNorm (wave-per-token). shifted!=0: dst is windowed order with
// cyclic shift applied (Swin partition); else identity row order.
// ======================================================================
__global__ __launch_bounds__(256)
void layernorm_kernel(const float* __restrict__ src,
                      const float* __restrict__ gamma,
                      const float* __restrict__ beta,
                      _Float16* __restrict__ dst, int shifted) {
    int lane = threadIdx.x & 31;
    int wave = threadIdx.x >> 5;
    int tw = blockIdx.x * 8 + wave;                 // destination row
    int srcRow = tw;
    if (shifted) {
        int win = tw >> 6, tok = tw & 63;
        int b = win >> 6, wh = (win >> 3) & 7, wwi = win & 7;
        int i = tok >> 3, j = tok & 7;
        int yh = (wh * 8 + i + SS_) & 63;
        int yw = (wwi * 8 + j + SS_) & 63;
        srcRow = (b * 64 + yh) * 64 + yw;
    }
    const float* p = src + (size_t)srcRow * DIMC + lane * 16;
    float4 xv[4];
    float s = 0.f;
    #pragma unroll
    for (int q = 0; q < 4; ++q) {
        xv[q] = ((const float4*)p)[q];
        s += xv[q].x + xv[q].y + xv[q].z + xv[q].w;
    }
    #pragma unroll
    for (int m = 16; m >= 1; m >>= 1) s += __shfl_xor(s, m, 32);
    float mean = s * (1.0f / DIMC);
    float v = 0.f;
    #pragma unroll
    for (int q = 0; q < 4; ++q) {
        float a0 = xv[q].x - mean, a1 = xv[q].y - mean;
        float a2 = xv[q].z - mean, a3 = xv[q].w - mean;
        v += a0 * a0 + a1 * a1 + a2 * a2 + a3 * a3;
    }
    #pragma unroll
    for (int m = 16; m >= 1; m >>= 1) v += __shfl_xor(v, m, 32);
    float rstd = rsqrtf(v * (1.0f / DIMC) + 1e-5f);

    const float4* gp = (const float4*)(gamma + lane * 16);
    const float4* bp = (const float4*)(beta  + lane * 16);
    v8h h0, h1;
    #pragma unroll
    for (int q = 0; q < 4; ++q) {
        float4 g4 = gp[q], b4 = bp[q];
        float o0 = (xv[q].x - mean) * rstd * g4.x + b4.x;
        float o1 = (xv[q].y - mean) * rstd * g4.y + b4.y;
        float o2 = (xv[q].z - mean) * rstd * g4.z + b4.z;
        float o3 = (xv[q].w - mean) * rstd * g4.w + b4.w;
        v8h* t = (q < 2) ? &h0 : &h1;
        int e = (q & 1) * 4;
        (*t)[e + 0] = (_Float16)o0; (*t)[e + 1] = (_Float16)o1;
        (*t)[e + 2] = (_Float16)o2; (*t)[e + 3] = (_Float16)o3;
    }
    _Float16* d = dst + (size_t)tw * DIMC + lane * 16;
    *(v8h*)d = h0;
    *(v8h*)(d + 8) = h1;
}

// ======================================================================
// QKV GEMM: [65536 x 512] f16  @  Wt[1536 x 512] -> scatter q/k/vT f16
// block tile 128(M) x 256(N), wave tile 32 x 128
// ======================================================================
__global__ __launch_bounds__(256)
void qkv_gemm_kernel(const _Float16* __restrict__ A,
                     const _Float16* __restrict__ Bt,
                     const float* __restrict__ biasv,
                     _Float16* __restrict__ qb,
                     _Float16* __restrict__ kb,
                     _Float16* __restrict__ vt) {
    int lane = threadIdx.x & 31, wave = threadIdx.x >> 5;
    int ln = lane & 15, lg = lane >> 4;
    int row0 = blockIdx.y * 128 + (wave >> 1) * 32;
    int col0 = blockIdx.x * 256 + (wave & 1) * 128;
    v8f acc[2][8] = {};
    gemm_core<8>(A, DIMC, Bt, DIMC, DIMC, row0, col0, lane, acc);

    #pragma unroll
    for (int mt = 0; mt < 2; ++mt)
    #pragma unroll
    for (int ct = 0; ct < 8; ++ct)
    #pragma unroll
    for (int r = 0; r < 8; ++r) {
        int row = row0 + mt * 16 + r + 8 * lg;
        int col = col0 + ct * 16 + ln;
        float val = acc[mt][ct][r] + biasv[col];
        int which = col >> 9, head = (col >> 7) & 3, d = col & 127;
        int win = row >> 6, tok = row & 63;
        size_t wh = (size_t)win * NHEAD + head;
        if (which == 0)
            qb[(wh * NTOK + tok) * HD + d] = (_Float16)(val * QSCALE);
        else if (which == 1)
            kb[(wh * NTOK + tok) * HD + d] = (_Float16)val;
        else
            vt[(wh * HD + d) * NTOK + tok] = (_Float16)val;
    }
}

// ======================================================================
// Attention: one block (4 waves) per (window, head).
// S = q k^T + bias; softmax; O = P v   (P staged in LDS as f16)
// ======================================================================
__global__ __launch_bounds__(128)
void attn_kernel(const _Float16* __restrict__ q,
                 const _Float16* __restrict__ k,
                 const _Float16* __restrict__ vT,
                 const float* __restrict__ biasmat,
                 _Float16* __restrict__ attnA) {
    __shared__ _Float16 Pl[NTOK * NTOK];
    int wh = blockIdx.x;                 // win*4 + head
    int win = wh >> 2, head = wh & 3;
    const _Float16* qbp = q  + (size_t)wh * NTOK * HD;
    const _Float16* kbp = k  + (size_t)wh * NTOK * HD;
    const _Float16* vbp = vT + (size_t)wh * HD * NTOK;
    int lane = threadIdx.x & 31, wave = threadIdx.x >> 5;
    int ln = lane & 15, lg = lane >> 4;
    int row0 = wave * 16;

    // ---- S = q @ k^T ----
    v8f s[4] = {};
    #pragma unroll
    for (int kt = 0; kt < 4; ++kt) {
        v16h a = load_a_frag(qbp, HD, row0, kt * 32, lane);
        #pragma unroll
        for (int ct = 0; ct < 4; ++ct) {
            v16h b = load_b_frag(kbp, HD, ct * 16, kt * 32, lane);
            s[ct] = wmma_f16(a, b, s[ct]);
        }
    }
    // ---- bias + softmax (per-row over 8 rows held by this lane) ----
    float pbuf[4][8];
    float rmax[8], rsum[8];
    #pragma unroll
    for (int r = 0; r < 8; ++r) {
        int row = row0 + r + 8 * lg;
        float m = -3.4e38f;
        #pragma unroll
        for (int ct = 0; ct < 4; ++ct) {
            float v = s[ct][r] + biasmat[row * 64 + ct * 16 + ln];
            pbuf[ct][r] = v;
            m = fmaxf(m, v);
        }
        rmax[r] = m;
    }
    #pragma unroll
    for (int r = 0; r < 8; ++r)
        #pragma unroll
        for (int m = 8; m >= 1; m >>= 1)
            rmax[r] = fmaxf(rmax[r], __shfl_xor(rmax[r], m, 16));
    #pragma unroll
    for (int r = 0; r < 8; ++r) {
        float su = 0.f;
        #pragma unroll
        for (int ct = 0; ct < 4; ++ct) {
            float e = __expf(pbuf[ct][r] - rmax[r]);
            pbuf[ct][r] = e;
            su += e;
        }
        rsum[r] = su;
    }
    #pragma unroll
    for (int r = 0; r < 8; ++r)
        #pragma unroll
        for (int m = 8; m >= 1; m >>= 1)
            rsum[r] += __shfl_xor(rsum[r], m, 16);
    #pragma unroll
    for (int r = 0; r < 8; ++r) {
        int row = row0 + r + 8 * lg;
        float inv = 1.0f / rsum[r];
        #pragma unroll
        for (int ct = 0; ct < 4; ++ct)
            Pl[row * 64 + ct * 16 + ln] = (_Float16)(pbuf[ct][r] * inv);
    }
    __syncthreads();

    // ---- O = P @ v ----
    #pragma unroll
    for (int dt = 0; dt < 8; ++dt) {
        v8f o = {};
        #pragma unroll
        for (int kt = 0; kt < 2; ++kt) {
            v16h a = load_a_frag(Pl, 64, row0, kt * 32, lane);   // ds loads
            v16h b = load_b_frag(vbp, NTOK, dt * 16, kt * 32, lane);
            o = wmma_f16(a, b, o);
        }
        #pragma unroll
        for (int r = 0; r < 8; ++r) {
            int row = row0 + r + 8 * lg;
            attnA[((size_t)win * NTOK + row) * DIMC + head * HD + dt * 16 + ln] =
                (_Float16)o[r];
        }
    }
}

// ======================================================================
// Proj GEMM + window-reverse + un-shift + residual -> x1 (f32)
// ======================================================================
__global__ __launch_bounds__(256)
void proj_gemm_kernel(const _Float16* __restrict__ A,
                      const _Float16* __restrict__ Bt,
                      const float* __restrict__ biasv,
                      const float* __restrict__ x,
                      float* __restrict__ x1) {
    int lane = threadIdx.x & 31, wave = threadIdx.x >> 5;
    int ln = lane & 15, lg = lane >> 4;
    int row0 = blockIdx.y * 128 + (wave >> 1) * 32;
    int col0 = blockIdx.x * 256 + (wave & 1) * 128;
    v8f acc[2][8] = {};
    gemm_core<8>(A, DIMC, Bt, DIMC, DIMC, row0, col0, lane, acc);

    #pragma unroll
    for (int mt = 0; mt < 2; ++mt)
    #pragma unroll
    for (int ct = 0; ct < 8; ++ct)
    #pragma unroll
    for (int r = 0; r < 8; ++r) {
        int row = row0 + mt * 16 + r + 8 * lg;
        int col = col0 + ct * 16 + ln;
        float val = acc[mt][ct][r] + biasv[col];
        int win = row >> 6, tok = row & 63;
        int b = win >> 6, whi = (win >> 3) & 7, wwi = win & 7;
        int i = tok >> 3, j = tok & 7;
        int yh = (whi * 8 + i + SS_) & 63;
        int yw = (wwi * 8 + j + SS_) & 63;
        size_t idx = (((size_t)b * 64 + yh) * 64 + yw) * DIMC + col;
        x1[idx] = x[idx] + val;
    }
}

// ======================================================================
// MLP1: h2 @ W1 + b1 -> exact GELU -> g (f16)
// ======================================================================
__global__ __launch_bounds__(256)
void mlp1_gemm_kernel(const _Float16* __restrict__ A,
                      const _Float16* __restrict__ Bt,
                      const float* __restrict__ biasv,
                      _Float16* __restrict__ g) {
    int lane = threadIdx.x & 31, wave = threadIdx.x >> 5;
    int ln = lane & 15, lg = lane >> 4;
    int row0 = blockIdx.y * 128 + (wave >> 1) * 32;
    int col0 = blockIdx.x * 256 + (wave & 1) * 128;
    v8f acc[2][8] = {};
    gemm_core<8>(A, DIMC, Bt, DIMC, DIMC, row0, col0, lane, acc);

    #pragma unroll
    for (int mt = 0; mt < 2; ++mt)
    #pragma unroll
    for (int ct = 0; ct < 8; ++ct)
    #pragma unroll
    for (int r = 0; r < 8; ++r) {
        int row = row0 + mt * 16 + r + 8 * lg;
        int col = col0 + ct * 16 + ln;
        float v = acc[mt][ct][r] + biasv[col];
        float gv = 0.5f * v * (1.0f + erff(v * 0.70710678118654752f));
        g[(size_t)row * 2048 + col] = (_Float16)gv;
    }
}

// ======================================================================
// MLP2: g @ W2 + b2 + x1 -> out (f32)
// ======================================================================
__global__ __launch_bounds__(256)
void mlp2_gemm_kernel(const _Float16* __restrict__ A,
                      const _Float16* __restrict__ Bt,
                      const float* __restrict__ biasv,
                      const float* __restrict__ x1,
                      float* __restrict__ out) {
    int lane = threadIdx.x & 31, wave = threadIdx.x >> 5;
    int ln = lane & 15, lg = lane >> 4;
    int row0 = blockIdx.y * 128 + (wave >> 1) * 32;
    int col0 = blockIdx.x * 256 + (wave & 1) * 128;
    v8f acc[2][8] = {};
    gemm_core<8>(A, 2048, Bt, 2048, 2048, row0, col0, lane, acc);

    #pragma unroll
    for (int mt = 0; mt < 2; ++mt)
    #pragma unroll
    for (int ct = 0; ct < 8; ++ct)
    #pragma unroll
    for (int r = 0; r < 8; ++r) {
        int row = row0 + mt * 16 + r + 8 * lg;
        int col = col0 + ct * 16 + ln;
        size_t idx = (size_t)row * DIMC + col;
        out[idx] = x1[idx] + acc[mt][ct][r] + biasv[col];
    }
}

// ======================================================================
// Launch
// ======================================================================
extern "C" void kernel_launch(void* const* d_in, const int* in_sizes, int n_in,
                              void* d_out, int out_size, void* d_ws, size_t ws_size,
                              hipStream_t stream) {
    (void)in_sizes; (void)n_in; (void)out_size; (void)ws_size;
    const float* x        = (const float*)d_in[0];
    const float* ln1_g    = (const float*)d_in[1];
    const float* ln1_b    = (const float*)d_in[2];
    const float* qkv_w    = (const float*)d_in[3];
    const float* qkv_b    = (const float*)d_in[4];
    const float* proj_w   = (const float*)d_in[5];
    const float* proj_b   = (const float*)d_in[6];
    const float* btab     = (const float*)d_in[7];
    const float* ln2_g    = (const float*)d_in[8];
    const float* ln2_b    = (const float*)d_in[9];
    const float* mlp_w1   = (const float*)d_in[10];
    const float* mlp_b1   = (const float*)d_in[11];
    const float* mlp_w2   = (const float*)d_in[12];
    const float* mlp_b2   = (const float*)d_in[13];
    float* out = (float*)d_out;

    // ---- workspace layout (bytes) ----
    char* w = (char*)d_ws;
    const size_t SZ_WQ  = (size_t)1536 * 512 * 2;
    const size_t SZ_WP  = (size_t)512  * 512 * 2;
    const size_t SZ_WM1 = (size_t)2048 * 512 * 2;
    const size_t SZ_WM2 = (size_t)2048 * 512 * 2;
    const size_t SZ_BM  = (size_t)64 * 64 * 4;
    const size_t SZ_ACT = (size_t)MROWS * DIMC * 2;   // 64 MiB
    size_t off = 0;
    _Float16* WQ   = (_Float16*)(w + off); off += SZ_WQ;
    _Float16* WP   = (_Float16*)(w + off); off += SZ_WP;
    _Float16* WM1  = (_Float16*)(w + off); off += SZ_WM1;
    _Float16* WM2  = (_Float16*)(w + off); off += SZ_WM2;
    float*    BM   = (float*)   (w + off); off += SZ_BM;
    off = (off + 255) & ~(size_t)255;
    _Float16* HWIN = (_Float16*)(w + off); off += SZ_ACT;  // also reused as h2
    char*     Rbeg = w + off;                               // 256 MiB region
    _Float16* QB   = (_Float16*)(Rbeg);
    _Float16* KB   = (_Float16*)(Rbeg + SZ_ACT);
    _Float16* VT   = (_Float16*)(Rbeg + 2 * SZ_ACT);
    _Float16* AA   = (_Float16*)(Rbeg + 3 * SZ_ACT);
    _Float16* G    = (_Float16*)(Rbeg);                    // reuse after proj
    off += 4 * SZ_ACT;
    float*    X1   = (float*)(w + off); off += (size_t)MROWS * DIMC * 4;

    // ---- prep: weight transposes (f32 -> f16^T) + bias matrix ----
    tpose_cvt_kernel<<<(1536 * 512) / 256, 256, 0, stream>>>(qkv_w,  WQ,  512, 1536);
    tpose_cvt_kernel<<<(512  * 512) / 256, 256, 0, stream>>>(proj_w, WP,  512, 512);
    tpose_cvt_kernel<<<(2048 * 512) / 256, 256, 0, stream>>>(mlp_w1, WM1, 512, 2048);
    tpose_cvt_kernel<<<(2048 * 512) / 256, 256, 0, stream>>>(mlp_w2, WM2, 2048, 512);
    biasmat_kernel<<<16, 256, 0, stream>>>(btab, BM);

    // ---- LN1 + shift + window partition ----
    layernorm_kernel<<<MROWS / 8, 256, 0, stream>>>(x, ln1_g, ln1_b, HWIN, 1);

    // ---- QKV GEMM ----
    qkv_gemm_kernel<<<dim3(1536 / 256, MROWS / 128), 256, 0, stream>>>(
        HWIN, WQ, qkv_b, QB, KB, VT);

    // ---- Attention (per window x head) ----
    attn_kernel<<<NWIN * NHEAD, 128, 0, stream>>>(QB, KB, VT, BM, AA);

    // ---- Proj + reverse shift + residual ----
    proj_gemm_kernel<<<dim3(512 / 256, MROWS / 128), 256, 0, stream>>>(
        AA, WP, proj_b, x, X1);

    // ---- LN2 (identity order) -> h2 in HWIN ----
    layernorm_kernel<<<MROWS / 8, 256, 0, stream>>>(X1, ln2_g, ln2_b, HWIN, 0);

    // ---- MLP ----
    mlp1_gemm_kernel<<<dim3(2048 / 256, MROWS / 128), 256, 0, stream>>>(
        HWIN, WM1, mlp_b1, G);
    mlp2_gemm_kernel<<<dim3(512 / 256, MROWS / 128), 256, 0, stream>>>(
        G, WM2, mlp_b2, X1, out);
}
